// AttConvReplacement_9002251453070
// MI455X (gfx1250) — matrix operand order repaired and checked by reading, not compiled, not run on hardware
//
#include <hip/hip_runtime.h>
#include <hip/hip_bf16.h>

// Problem constants (match the reference)
#define CC    64
#define NPIX  4096      // H*W
#define TOT   16384     // BS*NPIX
#define KSZ   7
#define RNB   49
#define NH    8
#define CHID  512       // positional FFN hidden width

typedef __attribute__((ext_vector_type(16))) __bf16 v16bf;
typedef __attribute__((ext_vector_type(8)))  float  v8f;

union BFrag { __bf16 h[16]; unsigned short s[16]; uint4 q[2]; v16bf v; };

// native f32 -> bf16 (RTNE) convert; compiler emits v_cvt (packs pairs)
__device__ __forceinline__ unsigned short f2bf(float f) {
  return __builtin_bit_cast(unsigned short, (__bf16)f);
}

__device__ __forceinline__ v8f wmma_bf16(v16bf a, v16bf b, v8f c) {
  // (neg_a, A, neg_b, B, c_mod, C, reuse_a, reuse_b)
  return __builtin_amdgcn_wmma_f32_16x16x32_bf16(false, a, false, b, (short)0, c,
                                                 false, false);
}

// ---------------------------------------------------------------------------
// Kernel 1: positional bias FFN.  P[p*49+rr][h] = relu(rel @ W1 + b1) @ W2 + b2
// M = n*r = 200704 tiled 16 rows/wave; K = 512 contracted with 16 bf16 WMMAs.
// Hidden activations are built per-lane directly in A-fragment layout.
// ---------------------------------------------------------------------------
__global__ __launch_bounds__(256) void pos_ffn_kernel(
    const float* __restrict__ pw1, const float* __restrict__ pb1,
    const float* __restrict__ pw2, const float* __restrict__ pb2,
    float* __restrict__ Pb)
{
  __shared__ float4 w1c[CHID];                     // {w1[0][k], w1[1][k], b1[k], 0}
  __shared__ __attribute__((aligned(16))) unsigned short W2t[16 * CHID]; // [n][k]
  __shared__ float  b2s[16];

  const int tid = threadIdx.x;
  for (int k = tid; k < CHID; k += 256)
    w1c[k] = make_float4(pw1[k], pw1[CHID + k], pb1[k], 0.0f);  // pw1 is [2][512]
  #pragma unroll
  for (int i = 0; i < 32; ++i) {                   // 16*512 entries
    int idx = tid + i * 256;
    int k = idx & (CHID - 1);
    int nn = idx >> 9;
    W2t[nn * CHID + k] = (nn < NH) ? f2bf(pw2[k * NH + nn]) : (unsigned short)0;
  }
  if (tid < 16) b2s[tid] = (tid < NH) ? pb2[tid] : 0.0f;
  __syncthreads();

  const int wave = tid >> 5;
  const int lane = tid & 31;
  const int m    = lane & 15;
  const int half = lane >> 4;
  const int tile = (blockIdx.x * 8 + wave) * 16;   // 1568*8*16 = 200704 exactly

  // relative coordinates for this lane's row (lanes l and l+16 share m)
  const int idx = tile + m;
  const int p  = idx / RNB;
  const int rr = idx - p * RNB;
  const int y = p >> 6, x = p & 63;
  const int dy = rr / KSZ - 3, dx = rr % KSZ - 3;
  const int ny = y + dy, nx = x + dx;
  const float step = 6.0f / 63.0f;
  const float gi = -3.0f + step * (float)y;
  const float gj = -3.0f + step * (float)x;
  const bool valid = ((unsigned)ny < 64u) && ((unsigned)nx < 64u);
  // zero-padded neighborhoods: OOB neighbor coord is (0,0)
  const float rel0 = (valid ? (-3.0f + step * (float)ny) : 0.0f) - gi;
  const float rel1 = (valid ? (-3.0f + step * (float)nx) : 0.0f) - gj;

  const int n = lane & 15;
  v8f acc = {};
  #pragma unroll
  for (int kt = 0; kt < 16; ++kt) {                // K = 512 = 16 * 32
    BFrag a;
    #pragma unroll
    for (int e = 0; e < 16; ++e) {
      // CDNA5 16-bit A layout: e<8 -> K=8*half+e ; e>=8 -> K=16+8*half+(e-8)
      int k = 32 * kt + ((e < 8) ? (8 * half + e) : (16 + 8 * half + (e - 8)));
      float4 w = w1c[k];                           // one ds_load_b128 (broadcast)
      float hv = fmaf(rel0, w.x, fmaf(rel1, w.y, w.z));
      a.h[e] = (__bf16)fmaxf(hv, 0.0f);            // native cvt, RTNE
    }
    BFrag b;                                       // B layout: same as A (e<8 -> K=8*half+e ; e>=8 -> K=16+8*half+(e-8))
    const uint4* bp = (const uint4*)&W2t[n * CHID + 32 * kt + 8 * half];
    b.q[0] = bp[0];
    b.q[1] = bp[2];
    acc = wmma_bf16(a.v, b.v, acc);
  }
  if (n < NH) {
    float bb = b2s[n];
    #pragma unroll
    for (int v = 0; v < 8; ++v) {
      int row = tile + 8 * half + v;
      Pb[row * NH + n] = acc[v] + bb;
    }
  }
}

// ---------------------------------------------------------------------------
// Kernel 2: fused Q/K/V projections.  X[16384][64] (built from NCHW x) @ W[64][64].
// 128 rows/block, 16 rows/wave, 3 proj * 4 ntiles * 2 ksteps = 24 WMMAs/wave.
// ---------------------------------------------------------------------------
__global__ __launch_bounds__(256) void qkv_proj_kernel(
    const float* __restrict__ x,
    const float* __restrict__ wq, const float* __restrict__ bq,
    const float* __restrict__ wk, const float* __restrict__ bk,
    const float* __restrict__ wv, const float* __restrict__ bv,
    float* __restrict__ Qb, float* __restrict__ Kb, float* __restrict__ Vb)
{
  __shared__ __attribute__((aligned(16))) unsigned short Xt[128 * 64];
  __shared__ __attribute__((aligned(16))) unsigned short Wt[3 * 64 * 64]; // [proj][n][k]
  __shared__ float bias_s[3 * 64];

  const int tid = threadIdx.x;
  #pragma unroll
  for (int i = 0; i < 48; ++i) {                   // 3*4096 weights, transposed
    int idx = tid + i * 256;
    int proj = idx >> 12;
    int rem  = idx & 4095;
    int n = rem & 63;
    int k = rem >> 6;
    const float* wsrc = (proj == 0) ? wq : (proj == 1) ? wk : wv;
    Wt[proj * 4096 + n * 64 + k] = f2bf(wsrc[k * 64 + n]);
  }
  if (tid < 192) {
    int proj = tid >> 6;
    const float* bsrc = (proj == 0) ? bq : (proj == 1) ? bk : bv;
    bias_s[tid] = bsrc[tid & 63];
  }
  const int rowbase = blockIdx.x * 128;
  const int bimg  = rowbase >> 12;                 // tile never straddles batches
  const int pbase = rowbase & 4095;
  #pragma unroll
  for (int i = 0; i < 32; ++i) {                   // NCHW -> [row][ch], coalesced in p
    int idx = tid + i * 256;
    int row = idx & 127;
    int ch  = idx >> 7;
    float v = x[bimg * (CC * NPIX) + ch * NPIX + pbase + row];
    Xt[row * 64 + ch] = f2bf(v);
  }
  __syncthreads();

  const int wave = tid >> 5;
  const int lane = tid & 31;
  const int m = lane & 15, half = lane >> 4;
  const int m0 = wave * 16;

  BFrag a0, a1;                                    // A frags for kt=0,1
  {
    const uint4* p0 = (const uint4*)&Xt[(m0 + m) * 64 + 8 * half];
    a0.q[0] = p0[0]; a0.q[1] = p0[2];              // runs at +0 and +16 elems
    const uint4* p1 = (const uint4*)&Xt[(m0 + m) * 64 + 32 + 8 * half];
    a1.q[0] = p1[0]; a1.q[1] = p1[2];
  }
  #pragma unroll
  for (int proj = 0; proj < 3; ++proj) {
    float* outb = (proj == 0) ? Qb : (proj == 1) ? Kb : Vb;
    const unsigned short* wbase = &Wt[proj * 4096];
    #pragma unroll
    for (int nt = 0; nt < 4; ++nt) {
      const int n = nt * 16 + (lane & 15);
      v8f c = {};
      #pragma unroll
      for (int kt = 0; kt < 2; ++kt) {
        BFrag bf;
        const uint4* bp = (const uint4*)&wbase[n * 64 + 32 * kt + 8 * half];
        bf.q[0] = bp[0]; bf.q[1] = bp[2];
        c = wmma_bf16((kt == 0) ? a0.v : a1.v, bf.v, c);
      }
      float bb = bias_s[proj * 64 + n];
      #pragma unroll
      for (int v = 0; v < 8; ++v) {
        int g = rowbase + m0 + 8 * half + v;
        outb[g * 64 + n] = c[v] + bb;
      }
    }
  }
}

// ---------------------------------------------------------------------------
// Kernel 3: attention core. One thread per (pixel, head); 8 consecutive
// threads cover one full 64-float K/V row -> coalesced. Scores live in VGPRs.
// V rows for pass 2 are prefetched (global_prefetch_b8) during pass 1.
// ---------------------------------------------------------------------------
__global__ __launch_bounds__(256) void attn_kernel(
    const float* __restrict__ Qb, const float* __restrict__ Kb,
    const float* __restrict__ Vb, const float* __restrict__ Pb,
    float* __restrict__ Ob)
{
  const int t = blockIdx.x * 256 + threadIdx.x;    // < TOT*NH
  const int head = t & 7;
  const int g = t >> 3;
  const int bimg = g >> 12;
  const int p = g & 4095;
  const int y = p >> 6, xx = p & 63;
  const float scale = 0.3535533905932738f;         // 1/sqrt(8)

  const float4* q4 = (const float4*)&Qb[g * 64 + head * 8];
  const float4 qa = q4[0], qh = q4[1];

  float sc[RNB];
  float mx = -3.4e38f;
  int rr = 0;
  #pragma unroll
  for (int dy = -3; dy <= 3; ++dy) {
    #pragma unroll
    for (int dx = -3; dx <= 3; ++dx, ++rr) {
      const int ny = y + dy, nx = xx + dx;
      const bool valid = ((unsigned)ny < 64u) && ((unsigned)nx < 64u);
      float s = 0.0f;                              // zero-padded keys -> dot = 0
      if (valid) {
        const int roff = ((bimg << 12) + (ny << 6) + nx) * 64 + head * 8;
        const float4* k4 = (const float4*)&Kb[roff];
        const float4 k0 = k4[0], k1 = k4[1];
        s = (qa.x * k0.x + qa.y * k0.y + qa.z * k0.z + qa.w * k0.w +
             qh.x * k1.x + qh.y * k1.y + qh.z * k1.z + qh.w * k1.w) * scale;
        __builtin_prefetch(&Vb[roff], 0, 3);       // warm V row for pass 2
      }
      s += Pb[(p * RNB + rr) * NH + head];
      sc[rr] = s;
      mx = fmaxf(mx, s);
    }
  }
  float sum = 0.0f;
  #pragma unroll
  for (int i = 0; i < RNB; ++i) {
    float e = __expf(sc[i] - mx);
    sc[i] = e;
    sum += e;
  }
  const float inv = 1.0f / sum;

  float4 acc0 = make_float4(0.f, 0.f, 0.f, 0.f);
  float4 acc1 = make_float4(0.f, 0.f, 0.f, 0.f);
  rr = 0;
  #pragma unroll
  for (int dy = -3; dy <= 3; ++dy) {
    #pragma unroll
    for (int dx = -3; dx <= 3; ++dx, ++rr) {
      const int ny = y + dy, nx = xx + dx;
      if (((unsigned)ny < 64u) && ((unsigned)nx < 64u)) {
        const float w = sc[rr] * inv;
        const float4* v4 =
            (const float4*)&Vb[((bimg << 12) + (ny << 6) + nx) * 64 + head * 8];
        const float4 v0 = v4[0], v1 = v4[1];
        acc0.x += w * v0.x; acc0.y += w * v0.y; acc0.z += w * v0.z; acc0.w += w * v0.w;
        acc1.x += w * v1.x; acc1.y += w * v1.y; acc1.z += w * v1.z; acc1.w += w * v1.w;
      }
    }
  }
  float4* o4 = (float4*)&Ob[g * 64 + head * 8];
  o4[0] = acc0; o4[1] = acc1;
}

// ---------------------------------------------------------------------------
// Kernel 4: output projection (WMMA), writes NCHW d_out directly.
// ---------------------------------------------------------------------------
__global__ __launch_bounds__(256) void out_proj_kernel(
    const float* __restrict__ Ob, const float* __restrict__ wo,
    const float* __restrict__ bo, float* __restrict__ out)
{
  __shared__ __attribute__((aligned(16))) unsigned short Xt[128 * 64];
  __shared__ __attribute__((aligned(16))) unsigned short Wt[64 * 64]; // [n][k]
  __shared__ float bias_s[64];

  const int tid = threadIdx.x;
  #pragma unroll
  for (int i = 0; i < 16; ++i) {
    int idx = tid + i * 256;
    int n = idx & 63, k = idx >> 6;
    Wt[n * 64 + k] = f2bf(wo[k * 64 + n]);
  }
  if (tid < 64) bias_s[tid] = bo[tid];
  const int rowbase = blockIdx.x * 128;
  #pragma unroll
  for (int i = 0; i < 32; ++i) {
    int idx = tid + i * 256;
    int ch = idx & 63, row = idx >> 6;
    Xt[row * 64 + ch] = f2bf(Ob[(rowbase + row) * 64 + ch]);
  }
  __syncthreads();

  const int wave = tid >> 5, lane = tid & 31;
  const int m = lane & 15, half = lane >> 4, m0 = wave * 16;

  BFrag a0, a1;
  const uint4* p0 = (const uint4*)&Xt[(m0 + m) * 64 + 8 * half];
  a0.q[0] = p0[0]; a0.q[1] = p0[2];
  const uint4* p1 = (const uint4*)&Xt[(m0 + m) * 64 + 32 + 8 * half];
  a1.q[0] = p1[0]; a1.q[1] = p1[2];

  #pragma unroll
  for (int nt = 0; nt < 4; ++nt) {
    const int n = nt * 16 + (lane & 15);
    v8f c = {};
    #pragma unroll
    for (int kt = 0; kt < 2; ++kt) {
      BFrag bf;
      const uint4* bp = (const uint4*)&Wt[n * 64 + 32 * kt + 8 * half];
      bf.q[0] = bp[0]; bf.q[1] = bp[2];
      c = wmma_bf16((kt == 0) ? a0.v : a1.v, bf.v, c);
    }
    float bb = bias_s[n];
    #pragma unroll
    for (int v = 0; v < 8; ++v) {
      int g = rowbase + m0 + 8 * half + v;
      int p = g & 4095;
      int bimg = g >> 12;
      out[bimg * (CC * NPIX) + n * NPIX + p] = c[v] + bb;  // NCHW
    }
  }
}

// ---------------------------------------------------------------------------
extern "C" void kernel_launch(void* const* d_in, const int* in_sizes, int n_in,
                              void* d_out, int out_size, void* d_ws, size_t ws_size,
                              hipStream_t stream) {
  (void)in_sizes; (void)n_in; (void)out_size; (void)ws_size;
  const float* x   = (const float*)d_in[0];
  const float* wq  = (const float*)d_in[1];
  const float* bq  = (const float*)d_in[2];
  const float* wk  = (const float*)d_in[3];
  const float* bk  = (const float*)d_in[4];
  const float* wv  = (const float*)d_in[5];
  const float* bv  = (const float*)d_in[6];
  const float* wo  = (const float*)d_in[7];
  const float* bo  = (const float*)d_in[8];
  const float* pw1 = (const float*)d_in[9];
  const float* pb1 = (const float*)d_in[10];
  const float* pw2 = (const float*)d_in[11];
  const float* pb2 = (const float*)d_in[12];
  float* out = (float*)d_out;

  // workspace layout (floats): Q | K | V | attnOut | P   (~23.2 MB total)
  float* Qb = (float*)d_ws;
  float* Kb = Qb + (size_t)TOT * CC;
  float* Vb = Kb + (size_t)TOT * CC;
  float* Ob = Vb + (size_t)TOT * CC;
  float* Pb = Ob + (size_t)TOT * CC;   // NPIX*RNB*NH floats

  pos_ffn_kernel<<<dim3(NPIX * RNB / 128), dim3(256), 0, stream>>>(pw1, pb1, pw2, pb2, Pb);
  qkv_proj_kernel<<<dim3(TOT / 128), dim3(256), 0, stream>>>(x, wq, bq, wk, bk, wv, bv,
                                                             Qb, Kb, Vb);
  attn_kernel<<<dim3(TOT * NH / 256), dim3(256), 0, stream>>>(Qb, Kb, Vb, Pb, Ob);
  out_proj_kernel<<<dim3(TOT / 128), dim3(256), 0, stream>>>(Ob, wo, bo, out);
}
